// TokenMixer_31928786879123
// MI455X (gfx1250) — compile-verified, matrix operand order checked by reading, not run on hardware
//
#include <hip/hip_runtime.h>
#include <hip/hip_bf16.h>
#include <math.h>

typedef __attribute__((ext_vector_type(16))) __bf16 v16bf;
typedef __attribute__((ext_vector_type(8)))  float  v8f;

#define DIMC 96
#define HEADS 3
#define HD 32
#define WS_ 8
#define SHIFT_ 4
#define NTOK 64
#define IMGH 256
#define IMGW 256
#define NB 8
#define HW (IMGH*IMGW)
#define NWIN 33
#define WINS_PER_IMG (NWIN*NWIN)
#define TOTAL_WIN (NB*WINS_PER_IMG)
#define ATT_SCALE 0.17677669529663687f  // 32^-0.5

__device__ __forceinline__ int refl(int i, int n) {
  if (i < 0) i = -i;
  if (i >= n) i = 2*n - 2 - i;
  return i;
}
// ISA 7.12.2: 16-bit A/B operand element e of v16bf in lane-half `hi`
// lane<16: K = {0..7, 16..23}; lane>=16: K = {8..15, 24..31}
__device__ __forceinline__ int kmap(int e, int hi) {
  return e + 8*hi + ((e >= 8) ? 8 : 0);
}

// ---------------------------------------------------------------- weights prep
__global__ void prep_weights(const float* __restrict__ conv_w,
                             const float* __restrict__ proj_w,
                             __bf16* __restrict__ wc, __bf16* __restrict__ wp) {
  int i = blockIdx.x*blockDim.x + threadIdx.x;
  if (i < DIMC*864) {              // wc[o][tap*96 + c] = conv_w[o][c][tap]
    int o = i / 864, r = i % 864;
    int tap = r / DIMC, c = r % DIMC;
    wc[i] = (__bf16)conv_w[(o*DIMC + c)*9 + tap];
  }
  int j = i - DIMC*864;
  if (j >= 0 && j < DIMC*DIMC) wp[j] = (__bf16)proj_w[j];
}

// ---------------------------------------------------------------- rel-pos bias
__global__ void bias_mlp(const float* __restrict__ m1_w, const float* __restrict__ m1_b,
                         const float* __restrict__ m2_w, const float* __restrict__ m2_b,
                         float* __restrict__ bias) {
  int i = blockIdx.x*blockDim.x + threadIdx.x;
  if (i >= NTOK*NTOK) return;
  int n = i / NTOK, m = i % NTOK;
  float di = (float)(n/WS_ - m/WS_);
  float dj = (float)(n%WS_ - m%WS_);
  float r0 = copysignf(log1pf(fabsf(di)), di);
  float r1 = copysignf(log1pf(fabsf(dj)), dj);
  float o0 = m2_b[0], o1 = m2_b[1], o2 = m2_b[2];
  for (int h = 0; h < 256; ++h) {
    float z = r0*m1_w[h*2+0] + r1*m1_w[h*2+1] + m1_b[h];
    float g = 0.5f*z*(1.0f + erff(z*0.70710678118f));   // exact gelu
    o0 += g*m2_w[0*256+h];
    o1 += g*m2_w[1*256+h];
    o2 += g*m2_w[2*256+h];
  }
  bias[0*4096+i] = o0; bias[1*4096+i] = o1; bias[2*4096+i] = o2;
}

// ---------------------------------------------------------------- QKV 1x1 GEMM
// Out = W(288x96) * X(96xP). One 16-pixel tile per wave, weights in LDS.
__global__ __launch_bounds__(256) void qkv_gemm(
    const float* __restrict__ xg,
    const float* __restrict__ Vw, const float* __restrict__ Vb,
    const float* __restrict__ QKw, const float* __restrict__ QKb,
    __bf16* __restrict__ Qo, __bf16* __restrict__ Ko, __bf16* __restrict__ Vo) {
  __shared__ __bf16 wlds[288*DIMC];
  __shared__ float  blds[288];
  int tid = threadIdx.x;
  for (int r = tid; r < 288*DIMC; r += 256) {
    int row = r / DIMC, c = r % DIMC;
    wlds[r] = (__bf16)((row < 192) ? QKw[row*DIMC + c] : Vw[(row-192)*DIMC + c]);
  }
  if (tid < 288) blds[tid] = (tid < 192) ? QKb[tid] : Vb[tid-192];
  __syncthreads();

  int wave = tid >> 5, lane = tid & 31;
  int hi = lane >> 4, c15 = lane & 15;
  long p = ((long)blockIdx.x*8 + wave)*16 + c15;          // this lane's pixel (B col)
  int b = (int)(p >> 16), idx = (int)(p & 65535);
  const float* xb = xg + (size_t)b*DIMC*HW + idx;

  v16bf Bv[3];                                            // X tile, K=96 in 3 steps
  #pragma unroll
  for (int s = 0; s < 3; ++s) {
    v16bf t;
    #pragma unroll
    for (int e = 0; e < 16; ++e) t[e] = (__bf16)xb[(size_t)(s*32 + kmap(e,hi))*HW];
    Bv[s] = t;
  }
  size_t p96 = (size_t)p * DIMC;
  for (int ot = 0; ot < 18; ++ot) {
    int arow = ot*16 + c15;
    v8f acc = {};
    #pragma unroll
    for (int s = 0; s < 3; ++s) {
      v16bf a;
      #pragma unroll
      for (int e = 0; e < 16; ++e) a[e] = wlds[arow*DIMC + s*32 + kmap(e,hi)];
      acc = __builtin_amdgcn_wmma_f32_16x16x32_bf16(false, a, false, Bv[s],
                                                    (short)0, acc, false, false);
    }
    int gr0 = ot*16;
    __bf16* dst; int cb;
    if      (gr0 <  96) { dst = Qo; cb = gr0;       }
    else if (gr0 < 192) { dst = Ko; cb = gr0 - 96;  }
    else                { dst = Vo; cb = gr0 - 192; }
    #pragma unroll
    for (int j = 0; j < 8; ++j) {
      int row = gr0 + hi*8 + j;
      dst[p96 + cb + hi*8 + j] = (__bf16)(acc[j] + blds[row]);
    }
  }
}

// ------------------------------------------------------------ window attention
// One wave per (window, head). S=QK^T (16 wmma), softmax, O=PV (16 wmma).
__global__ __launch_bounds__(128) void attn_win(
    const __bf16* __restrict__ Qg, const __bf16* __restrict__ Kg,
    const __bf16* __restrict__ Vg, const float* __restrict__ biasg,
    __bf16* __restrict__ att) {
  __shared__ __bf16 Pb[4][NTOK*NTOK];
  __shared__ int    tok[4][NTOK];
  int wave = threadIdx.x >> 5, lane = threadIdx.x & 31;
  int task = blockIdx.x*4 + wave;
  if (task >= TOTAL_WIN*HEADS) task = 0;                  // grid is exact; safety clamp
  int win = task / HEADS, head = task % HEADS;
  int b = win / WINS_PER_IMG, wr = win % WINS_PER_IMG;
  int wy = wr / NWIN, wx = wr % NWIN;

  for (int t = lane; t < NTOK; t += 32) {                 // reflect-padded token sources
    int sy = refl(wy*WS_ + (t >> 3) - SHIFT_, IMGH);
    int sx = refl(wx*WS_ + (t & 7)  - SHIFT_, IMGW);
    tok[wave][t] = b*HW + sy*IMGW + sx;
  }
  __syncthreads();

  int hi = lane >> 4, c15 = lane & 15;
  int hofs = head*HD;

  v16bf Bk[4];                                            // K col-tiles (reused by all rt)
  #pragma unroll
  for (int ct = 0; ct < 4; ++ct) {
    size_t base = (size_t)tok[wave][ct*16 + c15]*DIMC + hofs;
    v16bf t;
    #pragma unroll
    for (int e = 0; e < 16; ++e) t[e] = Kg[base + kmap(e,hi)];
    Bk[ct] = t;
  }
  for (int rt = 0; rt < 4; ++rt) {
    size_t qb = (size_t)tok[wave][rt*16 + c15]*DIMC + hofs;
    v16bf a;
    #pragma unroll
    for (int e = 0; e < 16; ++e) a[e] = Qg[qb + kmap(e,hi)];
    v8f s[4];
    #pragma unroll
    for (int ct = 0; ct < 4; ++ct) {
      v8f z = {};
      s[ct] = __builtin_amdgcn_wmma_f32_16x16x32_bf16(false, a, false, Bk[ct],
                                                      (short)0, z, false, false);
    }
    #pragma unroll
    for (int j = 0; j < 8; ++j) {                         // row softmax over 64 keys
      int qrow = rt*16 + hi*8 + j;
      const float* bp = biasg + head*4096 + qrow*64;
      float v0 = s[0][j]*ATT_SCALE + bp[ 0 + c15];
      float v1 = s[1][j]*ATT_SCALE + bp[16 + c15];
      float v2 = s[2][j]*ATT_SCALE + bp[32 + c15];
      float v3 = s[3][j]*ATT_SCALE + bp[48 + c15];
      float mx = fmaxf(fmaxf(v0,v1), fmaxf(v2,v3));
      #pragma unroll
      for (int d = 1; d < 16; d <<= 1) mx = fmaxf(mx, __shfl_xor(mx, d, 32));
      float e0 = __expf(v0-mx), e1 = __expf(v1-mx), e2 = __expf(v2-mx), e3 = __expf(v3-mx);
      float sm = e0+e1+e2+e3;
      #pragma unroll
      for (int d = 1; d < 16; d <<= 1) sm += __shfl_xor(sm, d, 32);
      float inv = 1.0f / sm;
      __bf16* pr = &Pb[wave][qrow*64];
      pr[ 0 + c15] = (__bf16)(e0*inv);
      pr[16 + c15] = (__bf16)(e1*inv);
      pr[32 + c15] = (__bf16)(e2*inv);
      pr[48 + c15] = (__bf16)(e3*inv);
    }
  }
  __syncthreads();

  v16bf Bv[2][2];                                         // V as B, reused across rt
  #pragma unroll
  for (int dc = 0; dc < 2; ++dc)
    #pragma unroll
    for (int ks = 0; ks < 2; ++ks) {
      v16bf t;
      #pragma unroll
      for (int e = 0; e < 16; ++e) {
        int tk = ks*32 + kmap(e,hi);
        t[e] = Vg[(size_t)tok[wave][tk]*DIMC + hofs + dc*16 + c15];
      }
      Bv[dc][ks] = t;
    }
  for (int rt = 0; rt < 4; ++rt) {
    v16bf a[2];
    #pragma unroll
    for (int ks = 0; ks < 2; ++ks) {
      v16bf t;
      #pragma unroll
      for (int e = 0; e < 16; ++e)
        t[e] = Pb[wave][(rt*16 + c15)*64 + ks*32 + kmap(e,hi)];
      a[ks] = t;
    }
    #pragma unroll
    for (int dc = 0; dc < 2; ++dc) {
      v8f acc = {};
      acc = __builtin_amdgcn_wmma_f32_16x16x32_bf16(false, a[0], false, Bv[dc][0],
                                                    (short)0, acc, false, false);
      acc = __builtin_amdgcn_wmma_f32_16x16x32_bf16(false, a[1], false, Bv[dc][1],
                                                    (short)0, acc, false, false);
      #pragma unroll
      for (int j = 0; j < 8; ++j) {
        int t  = rt*16 + hi*8 + j;
        int oy = wy*WS_ + (t >> 3) - SHIFT_;
        int ox = wx*WS_ + (t & 7)  - SHIFT_;
        if ((unsigned)oy < IMGH && (unsigned)ox < IMGW) {
          att[((size_t)b*HW + oy*IMGW + ox)*DIMC + hofs + dc*16 + c15] = (__bf16)acc[j];
        }
      }
    }
  }
}

// ---------------------------------------------- 3x3 reflect conv + attn -> mid
// Implicit GEMM: M=96 out-ch, K=864 (tap-major), N=pixels. 162 wmma / wave-tile.
__global__ __launch_bounds__(256) void conv_mix(
    const __bf16* __restrict__ Vg, const float* __restrict__ convb,
    const __bf16* __restrict__ wc, const __bf16* __restrict__ att,
    __bf16* __restrict__ mid) {
  int tid = threadIdx.x;
  int wave = tid >> 5, lane = tid & 31;
  int hi = lane >> 4, c15 = lane & 15;
  long p = ((long)blockIdx.x*8 + wave)*16 + c15;
  int b = (int)(p >> 16), idx = (int)(p & 65535);
  int y = idx >> 8, x = idx & 255;
  size_t tapb[9];
  #pragma unroll
  for (int t = 0; t < 9; ++t) {
    int yy = refl(y + t/3 - 1, IMGH);
    int xx = refl(x + t%3 - 1, IMGW);
    tapb[t] = ((size_t)b*HW + yy*IMGW + xx)*DIMC;
  }
  v8f acc[6] = {};
  for (int s = 0; s < 27; ++s) {
    int tap = s/3, c0 = (s%3)*32;
    v16bf bv;
    #pragma unroll
    for (int e = 0; e < 16; ++e) bv[e] = Vg[tapb[tap] + c0 + kmap(e,hi)];
    #pragma unroll
    for (int ot = 0; ot < 6; ++ot) {
      v16bf a;
      #pragma unroll
      for (int e = 0; e < 16; ++e) a[e] = wc[(ot*16 + c15)*864 + s*32 + kmap(e,hi)];
      acc[ot] = __builtin_amdgcn_wmma_f32_16x16x32_bf16(false, a, false, bv,
                                                        (short)0, acc[ot], false, false);
    }
  }
  size_t p96 = (size_t)p * DIMC;
  #pragma unroll
  for (int ot = 0; ot < 6; ++ot) {
    #pragma unroll
    for (int j = 0; j < 8; ++j) {
      int row = ot*16 + hi*8 + j;
      mid[p96 + row] = (__bf16)(acc[ot][j] + convb[row] + (float)att[p96 + row]);
    }
  }
}

// ------------------------------------------------------------- final 1x1 proj
__global__ __launch_bounds__(256) void proj_out(
    const __bf16* __restrict__ mid, const __bf16* __restrict__ wp,
    const float* __restrict__ pb, float* __restrict__ outg) {
  int tid = threadIdx.x;
  int wave = tid >> 5, lane = tid & 31;
  int hi = lane >> 4, c15 = lane & 15;
  long p = ((long)blockIdx.x*8 + wave)*16 + c15;
  int b = (int)(p >> 16), idx = (int)(p & 65535);
  size_t p96 = (size_t)p * DIMC;
  v16bf bv[3];
  #pragma unroll
  for (int s = 0; s < 3; ++s) {
    v16bf t;
    #pragma unroll
    for (int e = 0; e < 16; ++e) t[e] = mid[p96 + s*32 + kmap(e,hi)];
    bv[s] = t;
  }
  #pragma unroll
  for (int ot = 0; ot < 6; ++ot) {
    v8f acc = {};
    #pragma unroll
    for (int s = 0; s < 3; ++s) {
      v16bf a;
      #pragma unroll
      for (int e = 0; e < 16; ++e) a[e] = wp[(ot*16 + c15)*DIMC + s*32 + kmap(e,hi)];
      acc = __builtin_amdgcn_wmma_f32_16x16x32_bf16(false, a, false, bv[s],
                                                    (short)0, acc, false, false);
    }
    #pragma unroll
    for (int j = 0; j < 8; ++j) {
      int row = ot*16 + hi*8 + j;
      outg[((size_t)b*DIMC + row)*HW + idx] = acc[j] + pb[row];
    }
  }
}

extern "C" void kernel_launch(void* const* d_in, const int* in_sizes, int n_in,
                              void* d_out, int out_size, void* d_ws, size_t ws_size,
                              hipStream_t stream) {
  const float* x      = (const float*)d_in[0];
  const float* V_w    = (const float*)d_in[1];
  const float* V_b    = (const float*)d_in[2];
  const float* QK_w   = (const float*)d_in[3];
  const float* QK_b   = (const float*)d_in[4];
  const float* conv_w = (const float*)d_in[5];
  const float* conv_b = (const float*)d_in[6];
  const float* proj_w = (const float*)d_in[7];
  const float* proj_b = (const float*)d_in[8];
  const float* m1_w   = (const float*)d_in[9];
  const float* m1_b   = (const float*)d_in[10];
  const float* m2_w   = (const float*)d_in[11];
  const float* m2_b   = (const float*)d_in[12];
  float* out = (float*)d_out;

  char* ws = (char*)d_ws;
  const size_t SZ = (size_t)NB*HW*DIMC*sizeof(__bf16);   // 100,663,296 B per tensor
  __bf16* Q   = (__bf16*)(ws);
  __bf16* K   = (__bf16*)(ws + SZ);
  __bf16* V   = (__bf16*)(ws + 2*SZ);
  __bf16* ATT = (__bf16*)(ws + 3*SZ);
  float*  bias= (float*)(ws + 4*SZ);
  __bf16* wc  = (__bf16*)(ws + 4*SZ + 64*1024);
  __bf16* wp  = (__bf16*)(ws + 4*SZ + 64*1024 + (size_t)2*DIMC*864);
  __bf16* MID = Q;                                       // Q dead after attention

  prep_weights<<<(DIMC*864 + DIMC*DIMC + 255)/256, 256, 0, stream>>>(conv_w, proj_w, wc, wp);
  bias_mlp<<<16, 256, 0, stream>>>(m1_w, m1_b, m2_w, m2_b, bias);
  qkv_gemm<<<4096, 256, 0, stream>>>(x, V_w, V_b, QK_w, QK_b, Q, K, V);
  attn_win<<<(TOTAL_WIN*HEADS)/4, 128, 0, stream>>>(Q, K, V, bias, ATT);
  conv_mix<<<4096, 256, 0, stream>>>(V, conv_b, wc, ATT, MID);
  proj_out<<<4096, 256, 0, stream>>>(MID, wp, proj_b, out);
}
